// Attention_50946902065218
// MI455X (gfx1250) — matrix operand order repaired and checked by reading, not compiled, not run on hardware
//
#include <hip/hip_runtime.h>
#include <hip/hip_bf16.h>
#include <math.h>

// Problem sizes (fixed by the reference)
#define B_ 256
#define L_ 196
#define E_ 2048
#define D_ 1024
#define A_ 512

typedef __attribute__((ext_vector_type(16))) __bf16 v16bf;
typedef __attribute__((ext_vector_type(8)))  float  v8f;

#define TILE_M 64
#define KSTEP  32
#define NTILES (A_ / 16)          // 32 B-tiles across A
#define KTILES (E_ / KSTEP)       // 64 K-steps

// ---------------------------------------------------------------------------
// Kernel P1: swizzle We (f32 [E,A]) -> WeS (bf16 fragment-native layout)
// WeS element index = ((kt*NTILES + nt)*32 + lane)*16 + e
//   lane: l15 = N within tile, khalf selects K half; e: K = kt*32 + khalf*16 + e
// ---------------------------------------------------------------------------
__global__ __launch_bounds__(256)
void swizzle_we_kernel(const float* __restrict__ We, unsigned short* __restrict__ WeS) {
    const int g    = blockIdx.x * 256 + threadIdx.x;   // 65536 lanes-worth total
    const int ln   = g & 31;
    const int tile = g >> 5;                           // tile = kt*NTILES + nt
    const int nt   = tile & (NTILES - 1);
    const int kt   = tile >> 5;
    const int N    = nt * 16 + (ln & 15);
    const int Kb   = kt * KSTEP + (ln >> 4) * 16;
    union { v16bf v; uint4 q[2]; } o;
    #pragma unroll
    for (int e = 0; e < 16; ++e)
        o.v[e] = (__bf16)We[(size_t)(Kb + e) * A_ + N];
    uint4* dst = (uint4*)(WeS + (size_t)g * 16);       // 32B contiguous per thread
    dst[0] = o.q[0];
    dst[1] = o.q[1];
}

// ---------------------------------------------------------------------------
// Kernel A: att2[b][a] = decoder_hidden[b] . Wd[:,a] + bd[a]     (tiny GEMM)
// ---------------------------------------------------------------------------
__global__ __launch_bounds__(256)
void dec_proj_kernel(const float* __restrict__ dh, const float* __restrict__ Wd,
                     const float* __restrict__ bd, float* __restrict__ att2) {
    const int b = blockIdx.x;
    const int t = threadIdx.x;           // handles columns t and t+256
    const float* __restrict__ h = dh + (size_t)b * D_;
    float a0 = 0.f, a1 = 0.f;
    for (int k = 0; k < D_; ++k) {
        const float hk = h[k];
        a0 = fmaf(hk, Wd[(size_t)k * A_ + t], a0);
        a1 = fmaf(hk, Wd[(size_t)k * A_ + t + 256], a1);
    }
    att2[(size_t)b * A_ + t]       = a0 + bd[t];
    att2[(size_t)b * A_ + t + 256] = a1 + bd[t + 256];
}

// ---------------------------------------------------------------------------
// Kernel B: fused  scores[r] = wf . tanh(enc[r]@We + be + att2[b(r)]) + bf
// 512 threads = 16 waves; workgroup tile 64 rows x 512 cols, K-step 32.
// Barrier-free K-loop: A fragments straight from global f32 (+cvt),
// B fragments straight from pre-swizzled bf16 WeS in L2.
// ---------------------------------------------------------------------------
__global__ __launch_bounds__(512)
void attn_score_kernel(const float* __restrict__ enc,          // [B*L, E]
                       const unsigned short* __restrict__ WeS, // swizzled bf16
                       const float* __restrict__ be,           // [A]
                       const float* __restrict__ att2,         // [B, A]
                       const float* __restrict__ wf,           // [A]
                       const float* __restrict__ bfp,          // scalar
                       float* __restrict__ scores) {           // [B*L]
    __shared__ float score_lds[TILE_M];

    const int tid    = threadIdx.x;
    const int wave   = tid >> 5;
    const int lane   = tid & 31;
    const int l15    = lane & 15;
    const int khalf  = lane >> 4;              // 0: K-low half, 1: K-high half
    const int m_sub  = wave & 3;               // 4 M-subtiles of 16 rows
    const int nstrip = wave >> 2;              // 4 N-strips of 128 cols
    const int nt0    = nstrip * 8;             // first B-tile of this strip
    const long row_base = (long)blockIdx.x * TILE_M;

    if (tid < TILE_M) score_lds[tid] = 0.0f;
    __syncthreads();

    v8f acc[8];
    #pragma unroll
    for (int j = 0; j < 8; ++j) {
        #pragma unroll
        for (int i = 0; i < 8; ++i) acc[j][i] = 0.0f;
    }

    // per-lane A row pointer (M = lane&15 within subtile)
    const long arowg = row_base + m_sub * 16 + l15;
    const float* __restrict__ arow = enc + arowg * (long)E_;
    const int koff = khalf * 16;               // 0 or 16 floats

    #pragma unroll 2
    for (int kt = 0; kt < KTILES; ++kt) {
        const int kk = kt * KSTEP;
        // ---- A fragment: 16x32 bf16, contiguous K order, same as WeS ----
        //  e0..15: K = kk + khalf*16 + e (chunks c0..c3)
        const float4 c0 = *(const float4*)(arow + kk + koff);
        const float4 c1 = *(const float4*)(arow + kk + koff + 4);
        const float4 c2 = *(const float4*)(arow + kk + koff + 8);
        const float4 c3 = *(const float4*)(arow + kk + koff + 12);
        v16bf a;
        a[0]  = (__bf16)c0.x; a[1]  = (__bf16)c0.y; a[2]  = (__bf16)c0.z; a[3]  = (__bf16)c0.w;
        a[4]  = (__bf16)c1.x; a[5]  = (__bf16)c1.y; a[6]  = (__bf16)c1.z; a[7]  = (__bf16)c1.w;
        a[8]  = (__bf16)c2.x; a[9]  = (__bf16)c2.y; a[10] = (__bf16)c2.z; a[11] = (__bf16)c2.w;
        a[12] = (__bf16)c3.x; a[13] = (__bf16)c3.y; a[14] = (__bf16)c3.z; a[15] = (__bf16)c3.w;

        // ---- 8 B fragments direct from swizzled L2-resident WeS ----
        const unsigned short* __restrict__ bptr =
            WeS + ((size_t)(kt * NTILES + nt0) * 32 + lane) * 16;
        #pragma unroll
        for (int j = 0; j < 8; ++j) {
            union { uint4 q[2]; v16bf v; } bfr;
            const uint4* bq = (const uint4*)(bptr + (size_t)j * 512); // next tile = 512 ushorts
            bfr.q[0] = bq[0];
            bfr.q[1] = bq[1];
            acc[j] = __builtin_amdgcn_wmma_f32_16x16x32_bf16(
                false, a, false, bfr.v, (short)0, acc[j], false, false);
        }
    }

    // ---- epilogue: + be + att2[b], tanh, * wf, reduce over A ----
    const float bf0 = bfp[0];
    const float* __restrict__ at2row[8];
    #pragma unroll
    for (int v = 0; v < 8; ++v) {
        const long rv = row_base + m_sub * 16 + v + 8 * khalf;   // row of acc VGPR v
        at2row[v] = att2 + (size_t)(rv / L_) * A_;
    }
    float rowpart[8];
    #pragma unroll
    for (int v = 0; v < 8; ++v) rowpart[v] = 0.0f;

    #pragma unroll
    for (int j = 0; j < 8; ++j) {
        const int n = (nt0 + j) * 16 + l15;                      // N of acc[j]
        const float ben = be[n];
        const float wfn = wf[n];
        #pragma unroll
        for (int v = 0; v < 8; ++v) {
            const float val = acc[j][v] + ben + at2row[v][n];
            rowpart[v] = fmaf(tanhf(val), wfn, rowpart[v]);
        }
    }
    // reduce across the 16 lanes sharing a row (xor masks stay inside half-wave)
    #pragma unroll
    for (int v = 0; v < 8; ++v) {
        float s = rowpart[v];
        s += __shfl_xor(s, 1);
        s += __shfl_xor(s, 2);
        s += __shfl_xor(s, 4);
        s += __shfl_xor(s, 8);
        if (l15 == 0)
            atomicAdd(&score_lds[m_sub * 16 + v + 8 * khalf], s); // ds_add_f32
    }
    __syncthreads();
    if (tid < TILE_M)
        scores[row_base + tid] = score_lds[tid] + bf0;
}

// ---------------------------------------------------------------------------
// Kernel C: masked softmax over L per batch
// ---------------------------------------------------------------------------
__global__ __launch_bounds__(256)
void softmax_kernel(const float* __restrict__ scores, const int* __restrict__ mask,
                    float* __restrict__ alpha) {
    __shared__ float red[256];
    const int b = blockIdx.x, t = threadIdx.x;
    float x = -INFINITY;
    if (t < L_) {
        x = scores[(size_t)b * L_ + t];
        if (mask[(size_t)b * L_ + t] == 0) x = -1e9f;
    }
    red[t] = x; __syncthreads();
    for (int s = 128; s > 0; s >>= 1) {
        if (t < s) red[t] = fmaxf(red[t], red[t + s]);
        __syncthreads();
    }
    const float mx = red[0];
    __syncthreads();
    const float e = (t < L_) ? __expf(x - mx) : 0.0f;
    red[t] = e; __syncthreads();
    for (int s = 128; s > 0; s >>= 1) {
        if (t < s) red[t] += red[t + s];
        __syncthreads();
    }
    const float inv = 1.0f / red[0];
    if (t < L_) alpha[(size_t)b * L_ + t] = e * inv;
}

// ---------------------------------------------------------------------------
// Kernel D: awe[b][e] = sum_l alpha[b][l] * enc[b][l][e]   (memory bound)
// grid (B, 2), 256 threads, float4 per thread -> covers E=2048
// ---------------------------------------------------------------------------
__global__ __launch_bounds__(256)
void weighted_sum_kernel(const float* __restrict__ enc, const float* __restrict__ alpha,
                         float* __restrict__ awe) {
    const int b  = blockIdx.x;
    const int e0 = blockIdx.y * 1024 + threadIdx.x * 4;
    const float* __restrict__ ep = enc + (size_t)b * L_ * E_ + e0;
    const float* __restrict__ ap = alpha + (size_t)b * L_;
    float4 acc = make_float4(0.f, 0.f, 0.f, 0.f);
    for (int l = 0; l < L_; ++l) {
        const float a = ap[l];
        const float4 v = *(const float4*)(ep + (size_t)l * E_);
        acc.x = fmaf(a, v.x, acc.x);
        acc.y = fmaf(a, v.y, acc.y);
        acc.z = fmaf(a, v.z, acc.z);
        acc.w = fmaf(a, v.w, acc.w);
    }
    *(float4*)(awe + (size_t)b * E_ + e0) = acc;
}

// ---------------------------------------------------------------------------
// launch
// ---------------------------------------------------------------------------
extern "C" void kernel_launch(void* const* d_in, const int* in_sizes, int n_in,
                              void* d_out, int out_size, void* d_ws, size_t ws_size,
                              hipStream_t stream) {
    const float* enc  = (const float*)d_in[0];   // [B,L,E]
    const float* dh   = (const float*)d_in[1];   // [B,D]
    const int*   mask = (const int*)  d_in[2];   // [B,L]
    const float* We   = (const float*)d_in[3];   // [E,A]
    const float* be   = (const float*)d_in[4];   // [A]
    const float* Wd   = (const float*)d_in[5];   // [D,A]
    const float* bd   = (const float*)d_in[6];   // [A]
    const float* wf   = (const float*)d_in[7];   // [A]
    const float* bf   = (const float*)d_in[8];   // scalar

    float* awe   = (float*)d_out;                   // [B,E]
    float* alpha = (float*)d_out + (size_t)B_ * E_; // [B,L]

    float* ws     = (float*)d_ws;
    float* att2   = ws;                              // [B,A]  131072 floats
    float* scores = ws + (size_t)B_ * A_;            // [B*L]   50176 floats
    unsigned short* WeS = (unsigned short*)(ws + (size_t)B_ * A_ + (size_t)B_ * L_);
                                                     // [E*A] bf16 = 2MB (16B aligned)

    // 0) one-time weight swizzle to bf16 fragment layout (L2 resident)
    swizzle_we_kernel<<<(KTILES * NTILES * 32) / 256, 256, 0, stream>>>(We, WeS);
    // 1) decoder projection
    dec_proj_kernel<<<B_, 256, 0, stream>>>(dh, Wd, bd, att2);
    // 2) fused bf16-WMMA GEMM + bias + tanh + wf-dot  (B*L/64 = 784 blocks)
    attn_score_kernel<<<(B_ * L_) / TILE_M, 512, 0, stream>>>(
        enc, WeS, be, att2, wf, bf, scores);
    // 3) masked softmax over L
    softmax_kernel<<<B_, 256, 0, stream>>>(scores, mask, alpha);
    // 4) attention-weighted encoder sum
    weighted_sum_kernel<<<dim3(B_, 2), 256, 0, stream>>>(enc, alpha, awe);
}
